// DualScaledDotProductAttention_18056042512913
// MI455X (gfx1250) — matrix operand order repaired and checked by reading, not compiled, not run on hardware
//
#include <hip/hip_runtime.h>
#include <hip/hip_bf16.h>

typedef __attribute__((ext_vector_type(16))) __bf16 v16bf;
typedef __attribute__((ext_vector_type(8)))  float  v8f;
typedef __attribute__((ext_vector_type(4)))  int    v4i;

#define BATCH 64
#define LSEQ  1024
#define FDIM  128
#define NCHUNK (LSEQ / 32)
#define NEG_BIG (-1.0e9f)
#define VSTRIDE 40    // bf16 elems per Vt row (80B, 16B aligned, bank-staggered)
#define PSTRIDE 40    // bf16 elems per P-staging row
#define KROWB  272    // bytes per Kt row (256B data + 16B pad, 16B aligned)

#if __has_builtin(__builtin_amdgcn_global_load_async_to_lds_b128) && \
    __has_builtin(__builtin_amdgcn_s_wait_asynccnt)
#define USE_ASYNC 1
typedef __attribute__((address_space(1))) v4i* gv4i_p;  // global 128b granule
typedef __attribute__((address_space(3))) v4i* lv4i_p;  // LDS 128b granule
#else
#define USE_ASYNC 0
#endif

// ---------------- fp32 -> bf16 conversion (vector x4) ----------------
__global__ void cvt_bf16_kernel(const float* __restrict__ src,
                                __bf16* __restrict__ dst, int n4) {
  int i = blockIdx.x * blockDim.x + threadIdx.x;
  if (i >= n4) return;
  float4 v = ((const float4*)src)[i];
  union { __bf16 b[4]; uint2 u; } p;
  p.b[0] = (__bf16)v.x; p.b[1] = (__bf16)v.y;
  p.b[2] = (__bf16)v.z; p.b[3] = (__bf16)v.w;
  ((uint2*)dst)[i] = p.u;
}

// ---- operand loaders (row-major [own-idx][K] memory, CDNA5 WMMA layouts) ----
// A (16x32 bf16): lane<16 -> K {0..7,16..23}; lane>=16 -> K {8..15,24..31}
__device__ inline v16bf load_a16(const __bf16* rowbase, int hf) {
  const char* p = (const char*)rowbase + hf * 16;
  union { uint4 u[2]; v16bf v; } r;
  r.u[0] = *(const uint4*)(p);
  r.u[1] = *(const uint4*)(p + 32);
  return r.v;
}
// B (32x16, stored as [n][k]): lane<16 -> K {0..7,16..23}; lane>=16 -> K {8..15,24..31}
__device__ inline v16bf load_b16(const __bf16* rowbase, int hf) {
  const char* p = (const char*)rowbase + hf * 16;
  union { uint4 u[2]; v16bf v; } r;
  r.u[0] = *(const uint4*)(p);
  r.u[1] = *(const uint4*)(p + 32);
  return r.v;
}

__device__ inline v8f wmma_bf16(v16bf a, v16bf b, v8f c) {
  return __builtin_amdgcn_wmma_f32_16x16x32_bf16(false, a, false, b,
                                                 (short)0, c, false, false);
}

// ---------------- projection: Y[r,a] = (X[r,:]·W[a,:] + bias[a]) * scale ----
__global__ __launch_bounds__(128) void proj_kernel(
    const __bf16* __restrict__ X, const __bf16* __restrict__ W,
    const float* __restrict__ bias, __bf16* __restrict__ Y, float scale) {
  const int lane = threadIdx.x & 31;
  const int wave = threadIdx.x >> 5;
  const int hf   = lane >> 4;
  const int lr   = lane & 15;
  const long row0 = ((long)blockIdx.x * 4 + wave) * 16;

  v16bf a[4];
  {
    const __bf16* xrow = X + (row0 + lr) * FDIM;
#pragma unroll
    for (int c = 0; c < 4; ++c) a[c] = load_a16(xrow + c * 32, hf);
  }
#pragma unroll
  for (int t = 0; t < 8; ++t) {
    const __bf16* wrow = W + (t * 16 + lr) * FDIM;
    v16bf bb[4];
#pragma unroll
    for (int c = 0; c < 4; ++c) bb[c] = load_b16(wrow + c * 32, hf);
    v8f acc = {0.f,0.f,0.f,0.f,0.f,0.f,0.f,0.f};
#pragma unroll
    for (int c = 0; c < 4; ++c) acc = wmma_bf16(a[c], bb[c], acc);
    float bv = bias[t * 16 + lr];
#pragma unroll
    for (int i = 0; i < 8; ++i) {
      long r = row0 + i + 8 * hf;
      Y[r * FDIM + t * 16 + lr] = (__bf16)((acc[i] + bv) * scale);
    }
  }
}

// ---------------- flash attention pass ----------------
struct VReg { uint4 a, b; };
__device__ inline VReg load_vrow(const __bf16* V, long bbase, int row, int f0) {
  const __bf16* p = V + (bbase + row) * FDIM + f0;
  VReg r; r.a = *(const uint4*)p; r.b = *(const uint4*)(p + 8);
  return r;
}
__device__ inline void store_vt(__bf16* vt, VReg r, int mm, int f0) {
  union { uint4 u[2]; __bf16 e[16]; } t; t.u[0] = r.a; t.u[1] = r.b;
#pragma unroll
  for (int j = 0; j < 16; ++j) vt[(f0 + j) * VSTRIDE + mm] = t.e[j];
}

#if USE_ASYNC
// async-copy one 32x128 bf16 K tile (row-major, padded rows) into LDS;
// each thread moves two 16B granules -> 2 ASYNCcnt increments per wave.
__device__ inline void kt_async(const __bf16* K, char* ktbuf, long bbase,
                                int m0, int tid) {
#pragma unroll
  for (int q = 0; q < 2; ++q) {
    int g = tid + q * 256;
    int row = g >> 4, c = g & 15;
    const __bf16* src = K + (bbase + m0 + row) * FDIM + c * 8;
    char* dst = ktbuf + row * KROWB + c * 16;
    __builtin_amdgcn_global_load_async_to_lds_b128((gv4i_p)src, (lv4i_p)dst,
                                                   0, 0);
  }
}
#endif

// out[row,f] = sum_col softmax_col(Q[row]·K[col]) * V[col,f]
// mask_t==0: mask[b,row,col]; mask_t==1: mask[b,col,row]
__global__ __launch_bounds__(256) void attn_kernel(
    const __bf16* __restrict__ Q, const __bf16* __restrict__ K,
    const __bf16* __restrict__ V, const int* __restrict__ mask,
    float* __restrict__ out, int mask_t) {
  __shared__ __align__(16) __bf16 VtS[2][FDIM * VSTRIDE];   // V^T tiles [f][m]
  __shared__ __align__(16) __bf16 Pst[8 * 16 * PSTRIDE];    // P staging / wave
#if USE_ASYNC
  __shared__ __align__(16) char KtS[2][32 * KROWB];         // K tiles [m][a]
#endif

  const int tid  = threadIdx.x;
  const int lane = tid & 31;
  const int wave = tid >> 5;
  const int hf   = lane >> 4;
  const int lr   = lane & 15;
  const int b    = blockIdx.y;
  const int rowbase = (blockIdx.x * 8 + wave) * 16;
  const long bbase  = (long)b * LSEQ;
  const long qrow0  = bbase + rowbase;
  const int mm  = tid & 31;            // V-copy granule mapping
  const int f0v = (tid >> 5) * 16;

  v16bf qa[4];
  {
    const __bf16* qrow = Q + (qrow0 + lr) * FDIM;
#pragma unroll
    for (int c = 0; c < 4; ++c) qa[c] = load_a16(qrow + c * 32, hf);
  }

  v8f acc[8];
  float run_m[8], run_s[8];
#pragma unroll
  for (int t = 0; t < 8; ++t) acc[t] = (v8f){0.f,0.f,0.f,0.f,0.f,0.f,0.f,0.f};
#pragma unroll
  for (int i = 0; i < 8; ++i) { run_m[i] = -3.0e38f; run_s[i] = 0.f; }

  __bf16* myP = Pst + wave * 16 * PSTRIDE;

  // ---- pipeline prologue: V chunk0 -> Vt[0]; V chunk1 -> regs; async K 0,1 ----
  {
    VReg r0 = load_vrow(V, bbase, mm, f0v);
    store_vt(VtS[0], r0, mm, f0v);
  }
  VReg vreg = load_vrow(V, bbase, 32 + mm, f0v);
#if USE_ASYNC
  kt_async(K, KtS[0], bbase, 0, tid);
  kt_async(K, KtS[1], bbase, 32, tid);
  __builtin_amdgcn_s_wait_asynccnt(2);   // chunk0's copies complete (in-order)
#endif
  __syncthreads();

  for (int mc = 0; mc < NCHUNK; ++mc) {
    const int cur = mc & 1, nxt = cur ^ 1;
    const int m0 = mc * 32;

    if (mc + 1 < NCHUNK) {
      long pfi = mask_t
          ? (bbase + (m0 + 32 + lane)) * LSEQ + rowbase
          : (bbase + rowbase + lr) * LSEQ + (m0 + 32 + hf * 16);
      __builtin_prefetch(&mask[pfi], 0, 0);
    }

    // ---- scores: batch all 8 B-operand loads, then 8 WMMAs ----
    v16bf kb[8];
#if USE_ASYNC
    {
      const char* kt = (const char*)KtS[cur];
#pragma unroll
      for (int c = 0; c < 4; ++c) {
        kb[c]     = load_b16((const __bf16*)(kt + lr * KROWB + c * 64), hf);
        kb[4 + c] = load_b16((const __bf16*)(kt + (16 + lr) * KROWB + c * 64), hf);
      }
    }
#else
    {
      const __bf16* krow0 = K + (bbase + m0 + lr) * FDIM;
      const __bf16* krow1 = krow0 + 16 * FDIM;
#pragma unroll
      for (int c = 0; c < 4; ++c) {
        kb[c]     = load_b16(krow0 + c * 32, hf);
        kb[4 + c] = load_b16(krow1 + c * 32, hf);
      }
    }
#endif
    v8f s0 = {0.f,0.f,0.f,0.f,0.f,0.f,0.f,0.f};
    v8f s1 = {0.f,0.f,0.f,0.f,0.f,0.f,0.f,0.f};
#pragma unroll
    for (int c = 0; c < 4; ++c) {
      s0 = wmma_bf16(qa[c], kb[c], s0);
      s1 = wmma_bf16(qa[c], kb[4 + c], s1);
    }

    // ---- mask + online softmax over these 32 columns ----
    float p0[8], p1[8];
#pragma unroll
    for (int i = 0; i < 8; ++i) {
      int r_i = rowbase + i + 8 * hf;
      long i00, i01;
      if (mask_t) {
        i00 = (bbase + (m0 + lr)) * LSEQ + r_i;
        i01 = i00 + 16L * LSEQ;
      } else {
        i00 = (bbase + r_i) * LSEQ + (m0 + lr);
        i01 = i00 + 16;
      }
      float v0 = (mask[i00] == 0) ? NEG_BIG : s0[i];
      float v1 = (mask[i01] == 0) ? NEG_BIG : s1[i];
      float tm = fmaxf(v0, v1);
#pragma unroll
      for (int o = 1; o < 16; o <<= 1) tm = fmaxf(tm, __shfl_xor(tm, o, 32));
      float nm   = fmaxf(run_m[i], tm);
      float corr = __expf(run_m[i] - nm);
      run_m[i] = nm;
      v0 = __expf(v0 - nm);
      v1 = __expf(v1 - nm);
      float rs = v0 + v1;
#pragma unroll
      for (int o = 1; o < 16; o <<= 1) rs += __shfl_xor(rs, o, 32);
      run_s[i] = run_s[i] * corr + rs;
      p0[i] = v0; p1[i] = v1;
#pragma unroll
      for (int t = 0; t < 8; ++t) acc[t][i] = acc[t][i] * corr;
    }

    // ---- P tile -> LDS (C layout -> A layout), same wave, DS in-order ----
#pragma unroll
    for (int i = 0; i < 8; ++i) {
      int r = i + 8 * hf;
      myP[r * PSTRIDE + lr]      = (__bf16)p0[i];
      myP[r * PSTRIDE + 16 + lr] = (__bf16)p1[i];
    }
    v16bf pa = load_a16(myP + lr * PSTRIDE, hf);

    // ---- PV: batch 8 B-operand loads from Vt, then 8 WMMAs ----
    {
      const __bf16* vt = VtS[cur];
      v16bf vb[8];
#pragma unroll
      for (int t = 0; t < 8; ++t)
        vb[t] = load_b16(vt + (t * 16 + lr) * VSTRIDE, hf);
#pragma unroll
      for (int t = 0; t < 8; ++t) acc[t] = wmma_bf16(pa, vb[t], acc[t]);
    }

    // ---- pipeline: stage next V tile, fetch V for mc+2, finish async K ----
    if (mc + 1 < NCHUNK) {
      store_vt(VtS[nxt], vreg, mm, f0v);                 // vreg holds chunk mc+1
      if (mc + 2 < NCHUNK) vreg = load_vrow(V, bbase, (mc + 2) * 32 + mm, f0v);
#if USE_ASYNC
      __builtin_amdgcn_s_wait_asynccnt(0);               // Kt[nxt] ready
#endif
    }
    __syncthreads();
#if USE_ASYNC
    if (mc + 2 < NCHUNK) kt_async(K, KtS[cur], bbase, (mc + 2) * 32, tid);
#endif
  }

  // ---- normalize + store fp32 ----
#pragma unroll
  for (int t = 0; t < 8; ++t) {
#pragma unroll
    for (int i = 0; i < 8; ++i) {
      long r = qrow0 + i + 8 * hf;
      out[r * FDIM + t * 16 + lr] = acc[t][i] / run_s[i];
    }
  }
}

extern "C" void kernel_launch(void* const* d_in, const int* in_sizes, int n_in,
                              void* d_out, int out_size, void* d_ws, size_t ws_size,
                              hipStream_t stream) {
  const float* f1   = (const float*)d_in[0];
  const float* f2   = (const float*)d_in[1];
  const int*   mask = (const int*)d_in[2];
  const float* W1   = (const float*)d_in[3];
  const float* b1   = (const float*)d_in[4];
  const float* W2   = (const float*)d_in[5];
  const float* b2   = (const float*)d_in[6];

  float* out1 = (float*)d_out;                              // [B,L2,F1]
  float* out2 = out1 + (size_t)BATCH * LSEQ * FDIM;         // [B,L1,F2]

  char* ws = (char*)d_ws;
  const size_t FSZ = (size_t)BATCH * LSEQ * FDIM * sizeof(__bf16);  // 16 MB
  __bf16* f1bf = (__bf16*)(ws);
  __bf16* f2bf = (__bf16*)(ws + FSZ);
  __bf16* qbf  = (__bf16*)(ws + 2 * FSZ);
  __bf16* kbf  = (__bf16*)(ws + 3 * FSZ);
  __bf16* w1bf = (__bf16*)(ws + 4 * FSZ);
  __bf16* w2bf = (__bf16*)(ws + 4 * FSZ + 32768);

  int n4 = BATCH * LSEQ * FDIM / 4;
  cvt_bf16_kernel<<<(n4 + 255) / 256, 256, 0, stream>>>(f1, f1bf, n4);
  cvt_bf16_kernel<<<(n4 + 255) / 256, 256, 0, stream>>>(f2, f2bf, n4);
  int w4 = FDIM * FDIM / 4;
  cvt_bf16_kernel<<<(w4 + 255) / 256, 256, 0, stream>>>(W1, w1bf, w4);
  cvt_bf16_kernel<<<(w4 + 255) / 256, 256, 0, stream>>>(W2, w2bf, w4);

  const float inv_sqrt_a = 0.08838834764831845f;  // 1/sqrt(128), folded into q
  proj_kernel<<<dim3(1024, 1), 128, 0, stream>>>(f1bf, w1bf, b1, qbf, inv_sqrt_a);
  proj_kernel<<<dim3(1024, 1), 128, 0, stream>>>(f2bf, w2bf, b2, kbf, 1.0f);

  // out2 = softmax_rows(q k^T) f2   (rows = l)
  attn_kernel<<<dim3(8, BATCH), 256, 0, stream>>>(qbf, kbf, f2bf, mask, out2, 0);
  // out1 = softmax_rows(k q^T) f1   (rows = m, mask transposed)
  attn_kernel<<<dim3(8, BATCH), 256, 0, stream>>>(kbf, qbf, f1bf, mask, out1, 1);
}
